// DHT_Layer_28046136442970
// MI455X (gfx1250) — matrix operand order repaired and checked by reading, not compiled, not run on hardware
//
#include <hip/hip_runtime.h>
#include <hip/hip_bf16.h>
#include <stdint.h>

// ---------------------------------------------------------------------------
// DHT layer pipeline for MI455X (gfx1250, wave32, WMMA + TDM):
//   conv1x1(256->128)+BN+ReLU -> Hough voting (100 angles x 100 rho)
//   -> conv3x3(128->128)+BN+ReLU -> conv3x3(128->128)+BN+ReLU
// Convs: implicit GEMM on v_wmma_f32_16x16x32_bf16 (bf16 in, f32 acc).
// Weight tiles DMA'd into LDS by the Tensor Data Mover (tensor_load_to_lds)
// with D# padding producing a 40-bf16 LDS pitch (aligned b128 fragment reads).
// Intermediates kept bf16 to halve HBM traffic (23.3 TB/s roofline).
// ---------------------------------------------------------------------------

typedef __attribute__((ext_vector_type(16))) __bf16 v16bf;
typedef __attribute__((ext_vector_type(8)))  __bf16 v8bf;
typedef __attribute__((ext_vector_type(8)))  float  v8f;

#define DIMC 128
#define HH   100
#define WW   100
#define NB   16
#define EPSF 1e-5f

#if __has_builtin(__builtin_amdgcn_tensor_load_to_lds) && \
    __has_builtin(__builtin_amdgcn_s_wait_tensorcnt)
#define USE_TDM 1
#else
#define USE_TDM 0
#endif

#if USE_TDM
typedef unsigned int u32x4 __attribute__((ext_vector_type(4)));
typedef int          i32x4 __attribute__((ext_vector_type(4)));
typedef int          i32x8 __attribute__((ext_vector_type(8)));

// TDM 2D tile load: 128 rows x 32 bf16, global row stride = strideElems.
// D# pad: after every 16 DWORDs (one 32-bf16 row) insert 4 DWORDs -> LDS rows
// land on a 40-bf16 (80B, 16B-aligned) pitch.
__device__ __forceinline__ void tdm_load_a(const __bf16* gsrc, unsigned ldsOff,
                                           int strideElems) {
  unsigned long long ga = (unsigned long long)(size_t)gsrc;
  u32x4 g0;
  g0[0] = 1u;                                   // count=1 (valid), user mode
  g0[1] = ldsOff;                               // LDS byte address
  g0[2] = (unsigned)ga;                         // global_addr[31:0]
  g0[3] = (unsigned)((ga >> 32) & 0x01FFFFFFull) | (2u << 30);  // [56:32]|type=2
  i32x8 g1;
  g1[0] = (int)((1u << 16)     // data_size = 2B
              | (1u << 20)     // pad_enable
              | (3u << 22)     // pad_interval: 16 DWORDs
              | (3u << 25));   // pad_amount: 4 DWORDs
  g1[1] = (int)(32u << 16);    // tensor_dim0 = 32 (row length)
  g1[2] = (int)(128u << 16);   // tensor_dim1 = 128 (rows)
  g1[3] = (int)(32u << 16);    // tile_dim0 = 32
  g1[4] = 128;                 // tile_dim1 = 128 (tile_dim2 = 0)
  g1[5] = strideElems;         // tensor_dim0_stride (data_size units)
  g1[6] = 0;
  g1[7] = 0;
  i32x4 z4 = {0, 0, 0, 0};
  i32x8 z8 = {0, 0, 0, 0, 0, 0, 0, 0};
  // 6-arg form (amdgpu-toolchain clang): groups 0..3 + group-2/3 alt + cpol
  __builtin_amdgcn_tensor_load_to_lds(g0, g1, z4, z4, z8, 0);
}
#endif

// --------------------------- weight repack (fp32 -> bf16) -------------------
// 1x1: w[co][ci] -> q[co*256 + ci]   (identity layout, [t=0][co][ci])
__global__ __launch_bounds__(256) void k_cvt_w1(const float* __restrict__ w,
                                                __bf16* __restrict__ q) {
  int i = blockIdx.x * 256 + threadIdx.x;
  if (i < 256 * DIMC) q[i] = (__bf16)w[i];
}

// 3x3: w[co][ci][3][3] -> q[t*128*128 + co*128 + ci]  (k contiguous per row)
__global__ __launch_bounds__(256) void k_cvt_w3(const float* __restrict__ w,
                                                __bf16* __restrict__ q) {
  int i = blockIdx.x * 256 + threadIdx.x;
  if (i >= 9 * DIMC * DIMC) return;
  int t = i / (DIMC * DIMC);
  int rem = i - t * DIMC * DIMC;          // rem = co*128 + ci
  q[i] = (__bf16)w[rem * 9 + t];
}

// --------------------------- WMMA implicit-GEMM conv ------------------------
// Block: 256 threads = 8 waves. Wave w owns M-tile w (16 out chans) and four
// 16x16 accumulators covering 64 output pixels. K loop over TAPS*(CIN/32)
// slices of v_wmma_f32_16x16x32_bf16. A tile DMA'd by TDM (wave 0), B tile
// staged through VGPRs (zero padding + dtype conversion).
template <typename TIN, typename TOUT, int CIN, int TAPS>
__global__ __launch_bounds__(256) void k_conv_wmma(
    const TIN* __restrict__ in, const __bf16* __restrict__ wq,
    const float* __restrict__ bias, const float* __restrict__ gamma,
    const float* __restrict__ beta, TOUT* __restrict__ out) {
  __shared__ __align__(16) __bf16 aT[DIMC * 40];  // [co][k] pitch 40, 10 KB
  __shared__ __align__(16) __bf16 bT[64 * 40];    // [px][k] pitch 40,  5 KB

  const int tid  = threadIdx.x;
  const int lane = tid & 31;
  const int wave = tid >> 5;
  const int kb   = lane >> 4;
  const int l15  = lane & 15;

  const int x0 = blockIdx.x * 64;
  const int y  = blockIdx.y;
  const int nb = blockIdx.z;

  v8f acc[4] = {};

  for (int t = 0; t < TAPS; ++t) {
    const int dy = (TAPS == 9) ? (t / 3 - 1) : 0;
    const int dx = (TAPS == 9) ? (t % 3 - 1) : 0;
    const int yy = y + dy;
    for (int cb = 0; cb < CIN / 32; ++cb) {
      __syncthreads();  // previous slice's fragment reads are done

#if USE_TDM
      if (tid < 32) {   // one wave issues the weight-tile DMA (EXEC-agnostic)
        tdm_load_a(wq + (size_t)(t * DIMC) * CIN + cb * 32,
                   (unsigned)(size_t)aT, CIN);
      }
#else
      for (int i = tid; i < DIMC * 32; i += 256) {
        int kk = i & 31, co = i >> 5;
        aT[co * 40 + kk] = wq[(size_t)(t * DIMC + co) * CIN + cb * 32 + kk];
      }
#endif
      // B tile: 32 cin x 64 pixels of the shifted input row (overlaps TDM).
      #pragma unroll
      for (int i = tid; i < 32 * 64; i += 256) {
        int px = i & 63, kk = i >> 6;
        int xx = x0 + px + dx;
        float v = 0.f;
        if (yy >= 0 && yy < HH && xx >= 0 && xx < WW)
          v = (float)in[((size_t)(nb * CIN + cb * 32 + kk) * HH + yy) * WW + xx];
        bT[px * 40 + kk] = (__bf16)v;
      }
#if USE_TDM
      if (tid < 32) __builtin_amdgcn_s_wait_tensorcnt(0);
#endif
      __syncthreads();

      // A fragment (16x32): lane<16 holds K 0..7,16..23; lane>=16 8..15,24..31
      const __bf16* ap = &aT[(wave * 16 + l15) * 40];
      v8bf alo = *(const v8bf*)(ap + kb * 8);
      v8bf ahi = *(const v8bf*)(ap + 16 + kb * 8);
      v16bf av;
      #pragma unroll
      for (int e = 0; e < 8; ++e) { av[e] = alo[e]; av[e + 8] = ahi[e]; }

      // B fragments (32x16): lane<16 holds K 0..7,16..23; lane>=16 8..15,24..31
      #pragma unroll
      for (int nt = 0; nt < 4; ++nt) {
        const __bf16* bp = &bT[(nt * 16 + l15) * 40];
        v8bf bl = *(const v8bf*)(bp + kb * 8);
        v8bf bh = *(const v8bf*)(bp + 16 + kb * 8);
        v16bf bv;
        #pragma unroll
        for (int e = 0; e < 8; ++e) { bv[e] = bl[e]; bv[e + 8] = bh[e]; }
        acc[nt] = __builtin_amdgcn_wmma_f32_16x16x32_bf16(
            false, av, false, bv, (short)0, acc[nt], false, false);
      }
    }
  }

  // Fused epilogue: relu((acc + bias) * gamma/sqrt(1+eps) + beta)
  const float rs = rsqrtf(1.0f + EPSF);
  #pragma unroll
  for (int v = 0; v < 8; ++v) {
    const int co = wave * 16 + kb * 8 + v;   // C/D layout: M = v + 8*(lane>=16)
    const float sc = gamma[co] * rs;
    const float bt = beta[co];
    const float bs = bias[co];
    const size_t rowb = ((size_t)(nb * DIMC + co) * HH + y) * WW;
    #pragma unroll
    for (int nt = 0; nt < 4; ++nt) {
      const int xx = x0 + nt * 16 + l15;     // N column = lane&15
      if (xx < WW)
        out[rowb + xx] = (TOUT)fmaxf(0.f, (acc[nt][v] + bs) * sc + bt);
    }
  }
}

// ------------------------------- Hough voting -------------------------------
// Block = (angle a, 16-channel slice). LDS: 100 rho bins x 16 ch f32 (6.4 KB),
// ds_add_f32 accumulation. rintf == np.round (RTE).
__global__ __launch_bounds__(256) void k_dht(const __bf16* __restrict__ h1,
                                             __bf16* __restrict__ out) {
  __shared__ float acc[100 * 16];
  const int a = blockIdx.x;            // 0..99
  const int ncBase = blockIdx.y * 16;  // 0..2047 step 16
  const int tid = threadIdx.x;

  for (int i = tid; i < 1600; i += 256) acc[i] = 0.f;
  __syncthreads();

  // irho = (floor(sqrt(100^2+100^2)) + 1) / (numRho - 1) = 142/99
  const float inv_irho = 99.0f / 142.0f;
  const float th = (float)a * (3.14159265358979323846f / 100.0f);
  const float ct = cosf(th) * inv_irho;
  const float st = sinf(th) * inv_irho;

  const int ch = tid & 15;
  const __bf16* row = h1 + (size_t)(ncBase + ch) * (HH * WW);
  for (int p = tid >> 4; p < HH * WW; p += 16) {
    const int yq = p / WW;
    const int xq = p - yq * WW;
    int r = (int)rintf((float)(xq - WW / 2) * ct + (float)(yq - HH / 2) * st)
            + 50;
    r = r < 0 ? 0 : (r > 99 ? 99 : r);
    atomicAdd(&acc[r * 16 + ch], (float)row[p]);  // ds_add_f32
  }
  __syncthreads();

  for (int i = tid; i < 1600; i += 256) {
    const int c2 = i & 15, r = i >> 4;
    out[(size_t)(ncBase + c2) * (HH * WW) + a * 100 + r] = (__bf16)acc[i];
  }
}

// ------------------------------- launcher -----------------------------------
extern "C" void kernel_launch(void* const* d_in, const int* in_sizes, int n_in,
                              void* d_out, int out_size, void* d_ws,
                              size_t ws_size, hipStream_t stream) {
  const float* x   = (const float*)d_in[0];
  const float* w1  = (const float*)d_in[1];
  const float* b1  = (const float*)d_in[2];
  const float* g1  = (const float*)d_in[3];
  const float* be1 = (const float*)d_in[4];
  const float* w2  = (const float*)d_in[5];
  const float* b2  = (const float*)d_in[6];
  const float* g2  = (const float*)d_in[7];
  const float* be2 = (const float*)d_in[8];
  const float* w3  = (const float*)d_in[9];
  const float* b3  = (const float*)d_in[10];
  const float* g3  = (const float*)d_in[11];
  const float* be3 = (const float*)d_in[12];
  float* outp = (float*)d_out;

  char* ws = (char*)d_ws;
  size_t off = 0;
  auto carve = [&](size_t bytes) {
    void* p = ws + off;
    off += (bytes + 255) & ~(size_t)255;
    return p;
  };
  const size_t act = (size_t)NB * DIMC * HH * WW;
  __bf16* h1 = (__bf16*)carve(act * sizeof(__bf16));    // conv1 out
  __bf16* hd = (__bf16*)carve(act * sizeof(__bf16));    // dht out
  __bf16* h2 = (__bf16*)carve(act * sizeof(__bf16));    // conv2 out
  __bf16* q1 = (__bf16*)carve((size_t)256 * DIMC * sizeof(__bf16));
  __bf16* q2 = (__bf16*)carve((size_t)9 * DIMC * DIMC * sizeof(__bf16));
  __bf16* q3 = (__bf16*)carve((size_t)9 * DIMC * DIMC * sizeof(__bf16));

  k_cvt_w1<<<(256 * DIMC + 255) / 256, 256, 0, stream>>>(w1, q1);
  k_cvt_w3<<<(9 * DIMC * DIMC + 255) / 256, 256, 0, stream>>>(w2, q2);
  k_cvt_w3<<<(9 * DIMC * DIMC + 255) / 256, 256, 0, stream>>>(w3, q3);

  dim3 gconv((WW + 63) / 64, HH, NB);   // 2 x 100 x 16 blocks
  k_conv_wmma<float, __bf16, 256, 1>
      <<<gconv, 256, 0, stream>>>(x, q1, b1, g1, be1, h1);

  k_dht<<<dim3(100, (NB * DIMC) / 16), 256, 0, stream>>>(h1, hd);

  k_conv_wmma<__bf16, __bf16, DIMC, 9>
      <<<gconv, 256, 0, stream>>>(hd, q2, b2, g2, be2, h2);
  k_conv_wmma<__bf16, float, DIMC, 9>
      <<<gconv, 256, 0, stream>>>(h2, q3, b3, g3, be3, outp);
}
